// rNet_841813590140
// MI455X (gfx1250) — compile-verified, matrix operand order checked by reading, not run on hardware
//
#include <hip/hip_runtime.h>
#include <hip/hip_bf16.h>

// ---------------------------------------------------------------------------
// R-Net forward for MI455X (gfx1250), wave32 + WMMA 16x16x32 f16->f32,
// async global->LDS staging (ASYNCcnt) where the toolchain exposes it.
// P=256, Q=48, B=16, H=128, D=556 (padded to 576 for K%32==0).
// ---------------------------------------------------------------------------

typedef __attribute__((ext_vector_type(16))) _Float16 v16h;
typedef __attribute__((ext_vector_type(8)))  float    v8f;
typedef __attribute__((ext_vector_type(4)))  int      v4i;

#define WMMA_F16(a, b, c) \
  __builtin_amdgcn_wmma_f32_16x16x32_f16(false, (a), false, (b), (short)0, (c), false, false)

#if defined(__has_builtin)
#  if __has_builtin(__builtin_amdgcn_global_load_async_to_lds_b128)
#    define HAVE_ASYNC_LDS 1
#  endif
#endif

#if defined(HAVE_ASYNC_LDS)
#  if __has_builtin(__builtin_amdgcn_s_wait_asynccnt)
#    define WAIT_ASYNC(n) __builtin_amdgcn_s_wait_asynccnt(n)
#  else
#    define WAIT_ASYNC(n) asm volatile("s_wait_asynccnt " #n ::: "memory")
#  endif
#else
#  define WAIT_ASYNC(n) ((void)0)
#endif

// A/B tile load: 16x32 f16 tile from row-major [rows, ld] storage.
// Lane 0-15 -> row (lane&15), K-half selected by lane>>4 (16 f16 each).
__device__ __forceinline__ v16h ld_tile(const _Float16* p, int ld, int lane) {
  return *(const v16h*)(p + (size_t)(lane & 15) * ld + ((lane >> 4) << 4));
}

// C/D store: VGPR i holds M = i + 8*(lane>>4), N = lane&15.
__device__ __forceinline__ void st_tile_f32(float* p, int ld, int lane, v8f c) {
  int col = lane & 15, r0 = (lane >> 4) * 8;
#pragma unroll
  for (int i = 0; i < 8; ++i) p[(size_t)(r0 + i) * ld + col] = c[i];
}
__device__ __forceinline__ void st_tile_f16(_Float16* p, int ld, int lane, v8f c) {
  int col = lane & 15, r0 = (lane >> 4) * 8;
#pragma unroll
  for (int i = 0; i < 8; ++i) p[(size_t)(r0 + i) * ld + col] = (_Float16)c[i];
}

__device__ __forceinline__ float sigm(float x) { return 1.0f / (1.0f + expf(-x)); }

// Stage one 16x32 f16 tile (1 KB) global->LDS: 64 lanes x 16 B.
// Async (ASYNCcnt-tracked) when available, else vectorized b128 copy.
__device__ __forceinline__ void stage_a_tile(_Float16* dst, const _Float16* src,
                                             int lda, int tid) {
  if (tid < 64) {                       // waves 0,1: uniform per wave, EXEC full
    int r = tid >> 2, c4 = (tid & 3) * 8;   // row, 8-f16 (16 B) chunk
    const _Float16* g = src + (size_t)r * lda + c4;
    _Float16* l = dst + r * 32 + c4;
#if defined(HAVE_ASYNC_LDS)
    __builtin_amdgcn_global_load_async_to_lds_b128(
        (__attribute__((address_space(1))) v4i*)(void*)g,
        (__attribute__((address_space(3))) v4i*)(void*)l, 0, 0);
#else
    *(uint4*)l = *(const uint4*)g;
#endif
  }
}

// ---------------------------------------------------------------------------
// f32 -> f16 convert with K-padding (zero fill).
// ---------------------------------------------------------------------------
__global__ void k_cast_pad(const float* __restrict__ src, _Float16* __restrict__ dst,
                           int rows, int K, int Kpad) {
  size_t n = (size_t)rows * Kpad;
  for (size_t i = (size_t)blockIdx.x * blockDim.x + threadIdx.x; i < n;
       i += (size_t)gridDim.x * blockDim.x) {
    int r = (int)(i / Kpad), c = (int)(i % Kpad);
    dst[i] = (c < K) ? (_Float16)src[(size_t)r * K + c] : (_Float16)0.0f;
  }
}

// ---------------------------------------------------------------------------
// Generic time-batched GEMM: C[M,N] = A[M,K] * B[N,K]^T + bias.
// grid = (N/128, M/16); 8 waves, one 16x16 C tile per wave.
// A chunks double-buffered in LDS via async global->LDS copies so the
// next chunk's DMA overlaps the current chunk's WMMA.
// ---------------------------------------------------------------------------
__global__ __launch_bounds__(256) void k_gemm_wmma(
    const _Float16* __restrict__ A, int lda,
    const _Float16* __restrict__ Bm, int ldb,
    const float* __restrict__ bias,
    float* __restrict__ C, _Float16* __restrict__ C16, int ldc, int K) {
  __shared__ __align__(64) _Float16 sA[2][16 * 32];
  int tid = threadIdx.x, lane = tid & 31, wave = tid >> 5;
  int mt = blockIdx.y;
  int nt = blockIdx.x * 8 + wave;
  const _Float16* Arow = A + (size_t)mt * 16 * lda;
  const int nk = K >> 5;

  stage_a_tile(sA[0], Arow, lda, tid);
  v8f acc = {};
  for (int i = 0; i < nk; ++i) {
    int kc = i << 5;
    if (i + 1 < nk) {
      stage_a_tile(sA[(i + 1) & 1], Arow + kc + 32, lda, tid);
      WAIT_ASYNC(1);                    // previous chunk landed; newest in flight
    } else {
      WAIT_ASYNC(0);
    }
    __syncthreads();
    v16h a = ld_tile(sA[i & 1], 32, lane);
    v16h b = ld_tile(Bm + (size_t)nt * 16 * ldb + kc, ldb, lane);
    acc = WMMA_F16(a, b, acc);
    __syncthreads();
  }
  int col = nt * 16 + (lane & 15), r0 = (lane >> 4) * 8;
#pragma unroll
  for (int i = 0; i < 8; ++i) {
    float v = acc[i] + (bias ? bias[col] : 0.0f);
    size_t o = (size_t)(mt * 16 + r0 + i) * ldc + col;
    C[o] = v;
    if (C16) C16[o] = (_Float16)v;
  }
}

// ---------------------------------------------------------------------------
// Recurrent 3-layer GRU, one (direction) per block; layer-0 input gates are
// precomputed (Gi0, includes bih0). Per step/layer: gh = h*Whh^T (WMMA),
// gi = x*Wih^T (WMMA, layers 1-2), then fused gate elementwise in fp32.
// N=384 -> 24 tiles, 3 per wave. LDS: 56 KB.
// ---------------------------------------------------------------------------
__global__ __launch_bounds__(256) void k_gru_dir(
    const float* __restrict__ Gi0, const float* __restrict__ bhh0,
    const _Float16* __restrict__ Whh0,
    const _Float16* __restrict__ Wih1, const _Float16* __restrict__ Whh1,
    const float* __restrict__ bih1, const float* __restrict__ bhh1,
    const _Float16* __restrict__ Wih2, const _Float16* __restrict__ Whh2,
    const float* __restrict__ bih2, const float* __restrict__ bhh2,
    float* __restrict__ Hout, _Float16* __restrict__ Hout16, int T) {
  __shared__ __align__(64) float    sH[3][16 * 128];   // fp32 master hidden states
  __shared__ __align__(64) _Float16 sAh[16 * 128];     // f16 A-operand (h_l, t-1)
  __shared__ __align__(64) _Float16 sAx[16 * 128];     // f16 A-operand (h_{l-1}, t)
  __shared__ __align__(64) _Float16 sGi[16 * 384];     // input-gate preacts
  __shared__ __align__(64) _Float16 sGh[16 * 384];     // hidden-gate preacts
  int tid = threadIdx.x, lane = tid & 31, wave = tid >> 5;
  int dir = blockIdx.x;

  for (int i = tid; i < 3 * 2048; i += 256) (&sH[0][0])[i] = 0.0f;
  __syncthreads();

  const _Float16* WihL[3] = {nullptr, Wih1, Wih2};
  const _Float16* WhhL[3] = {Whh0, Whh1, Whh2};
  const float*    bihL[3] = {nullptr, bih1, bih2};
  const float*    bhhL[3] = {bhh0, bhh1, bhh2};

  for (int t = 0; t < T; ++t) {
    int tt = dir ? (T - 1 - t) : t;
    // pull next step's layer-0 gates toward the WGP while this step computes
    if (t + 1 < T) {
      int tn = dir ? (T - 2 - t) : (t + 1);
      __builtin_prefetch(Gi0 + (size_t)tn * 16 * 384 + (size_t)tid * 24, 0, 1);
    }
    for (int l = 0; l < 3; ++l) {
      for (int i = tid; i < 2048; i += 256) {
        sAh[i] = (_Float16)sH[l][i];
        if (l > 0) sAx[i] = (_Float16)sH[l - 1][i];
      }
      __syncthreads();
#pragma unroll
      for (int j = 0; j < 3; ++j) {
        int nt = wave * 3 + j;
        v8f acch = {}, acci = {};
#pragma unroll
        for (int kc = 0; kc < 128; kc += 32) {
          v16h a  = ld_tile(sAh + kc, 128, lane);
          v16h bh = ld_tile(WhhL[l] + (size_t)nt * 16 * 128 + kc, 128, lane);
          acch = WMMA_F16(a, bh, acch);
          if (l > 0) {
            v16h ax = ld_tile(sAx + kc, 128, lane);
            v16h bi = ld_tile(WihL[l] + (size_t)nt * 16 * 128 + kc, 128, lane);
            acci = WMMA_F16(ax, bi, acci);
          }
        }
        st_tile_f16(sGh + nt * 16, 384, lane, acch);
        if (l > 0) st_tile_f16(sGi + nt * 16, 384, lane, acci);
      }
      __syncthreads();
      for (int e = tid; e < 2048; e += 256) {
        int bi = e >> 7, c = e & 127;
        float gr, gz, gn;
        if (l == 0) {
          const float* g = Gi0 + (size_t)tt * 16 * 384 + (size_t)bi * 384;
          gr = g[c]; gz = g[c + 128]; gn = g[c + 256];
        } else {
          const float* bb = bihL[l];
          gr = (float)sGi[bi * 384 + c] + bb[c];
          gz = (float)sGi[bi * 384 + c + 128] + bb[c + 128];
          gn = (float)sGi[bi * 384 + c + 256] + bb[c + 256];
        }
        const float* bh = bhhL[l];
        float hr = (float)sGh[bi * 384 + c] + bh[c];
        float hz = (float)sGh[bi * 384 + c + 128] + bh[c + 128];
        float hn = (float)sGh[bi * 384 + c + 256] + bh[c + 256];
        float r = sigm(gr + hr);
        float z = sigm(gz + hz);
        float n = tanhf(gn + r * hn);
        float h2 = (1.0f - z) * n + z * sH[l][bi * 128 + c];
        sH[l][bi * 128 + c] = h2;
        if (l == 2) {
          size_t o = (size_t)tt * 16 * 256 + (size_t)bi * 256 + dir * 128 + c;
          Hout[o] = h2;
          Hout16[o] = (_Float16)h2;
        }
      }
      __syncthreads();
    }
  }
}

// ---------------------------------------------------------------------------
// Gated additive-attention LSTM (match-LSTM / self-matching), one direction
// per block. WMMA for x*Wcur^T + h*Whid^T (N=128), z*Wg^T, z*Wih^T, h*Whh^T
// (N=512, 4 tiles/wave, K up to 512). Attention scores / softmax / weighted
// sum in fp32 VALU. LDS: 64 KB.
// ---------------------------------------------------------------------------
__global__ __launch_bounds__(256) void k_attn_lstm(
    const _Float16* __restrict__ Hseq16,  // [T,B,256]
    const float* __restrict__ Hmem,       // [Tm,B,256]
    const float* __restrict__ AttMem,     // [Tm*B,128] = Hmem @ Wmem^T
    const _Float16* __restrict__ Wcur,    // [128,256]
    const _Float16* __restrict__ Whid,    // [128,128]
    const float* __restrict__ w_att,      // [128]
    const _Float16* __restrict__ Wg,      // [512,512]
    const _Float16* __restrict__ Wih,     // [512,512]
    const _Float16* __restrict__ Whh,     // [512,128]
    const float* __restrict__ bias,       // [512]
    float* __restrict__ Hout, _Float16* __restrict__ Hout16,
    int T, int Tm) {
  __shared__ __align__(64) float    sh[16 * 128];
  __shared__ __align__(64) float    sc_[16 * 128];
  __shared__ __align__(64) _Float16 sZ[16 * 512];   // f16 A staging (h or z)
  __shared__ __align__(64) float    sScr[16 * 512]; // proj[0:2048] | scores[2048:] | gates
  int tid = threadIdx.x, lane = tid & 31, wave = tid >> 5;
  int dir = blockIdx.x;

  for (int i = tid; i < 2048; i += 256) { sh[i] = 0.0f; sc_[i] = 0.0f; }
  __syncthreads();

  for (int t = 0; t < T; ++t) {
    int tt = dir ? (T - 1 - t) : t;
    if (t + 1 < T) {  // prefetch next step's x_t rows
      int tn = dir ? (T - 2 - t) : (t + 1);
      __builtin_prefetch(Hseq16 + (size_t)tn * 4096 + (size_t)tid * 16, 0, 1);
    }

    // -- phase 1: stage h as f16 ([16,128] at sZ[0:2048])
    for (int i = tid; i < 2048; i += 256) sZ[i] = (_Float16)sh[i];
    __syncthreads();

    // -- phase 2: proj = x_t*Wcur^T + h*Whid^T  (N=128 -> 1 tile/wave)
    {
      v8f acc = {};
#pragma unroll
      for (int kc = 0; kc < 256; kc += 32) {
        v16h a = ld_tile(Hseq16 + (size_t)tt * 4096 + kc, 256, lane);
        v16h b = ld_tile(Wcur + (size_t)wave * 16 * 256 + kc, 256, lane);
        acc = WMMA_F16(a, b, acc);
      }
#pragma unroll
      for (int kc = 0; kc < 128; kc += 32) {
        v16h a = ld_tile(sZ + kc, 128, lane);
        v16h b = ld_tile(Whid + (size_t)wave * 16 * 128 + kc, 128, lane);
        acc = WMMA_F16(a, b, acc);
      }
      st_tile_f32(sScr + wave * 16, 128, lane, acc);
    }
    __syncthreads();

    // -- phase 3: attention scores s[q,b] = w_att . tanh(att_mem + proj)
    for (int e = tid; e < Tm * 16; e += 256) {
      int b = e & 15;
      const float* am = AttMem + (size_t)e * 128;
      const float* pr = sScr + b * 128;
      float s = 0.0f;
      for (int c = 0; c < 128; ++c) s += w_att[c] * tanhf(am[c] + pr[c]);
      sScr[2048 + e] = s;
    }
    __syncthreads();

    // -- phase 4: softmax over memory positions, per batch element
    if (tid < 16) {
      int b = tid;
      float m = -1e30f;
      for (int q = 0; q < Tm; ++q) m = fmaxf(m, sScr[2048 + q * 16 + b]);
      float s = 0.0f;
      for (int q = 0; q < Tm; ++q) {
        float ex = expf(sScr[2048 + q * 16 + b] - m);
        sScr[2048 + q * 16 + b] = ex;
        s += ex;
      }
      float inv = 1.0f / s;
      for (int q = 0; q < Tm; ++q) sScr[2048 + q * 16 + b] *= inv;
    }
    __syncthreads();

    // -- phase 5: weighted memory + z = [x_t, w] as f16 ([16,512])
    for (int e = tid; e < 4096; e += 256) {
      int b = e >> 8, d = e & 255;
      float wv = 0.0f;
      for (int q = 0; q < Tm; ++q)
        wv += sScr[2048 + q * 16 + b] * Hmem[((size_t)q * 16 + b) * 256 + d];
      sZ[b * 512 + 256 + d] = (_Float16)wv;
      sZ[b * 512 + d] = Hseq16[(size_t)tt * 4096 + (size_t)b * 256 + d];
    }
    __syncthreads();

    // -- phase 6: gate z <- z * sigmoid(z*Wg^T)  (N=512 -> 4 tiles/wave)
    {
      v8f accg[4];
#pragma unroll
      for (int j = 0; j < 4; ++j) {
        int nt = wave * 4 + j;
        accg[j] = (v8f){};
#pragma unroll
        for (int kc = 0; kc < 512; kc += 32) {
          v16h a = ld_tile(sZ + kc, 512, lane);
          v16h b = ld_tile(Wg + (size_t)nt * 16 * 512 + kc, 512, lane);
          accg[j] = WMMA_F16(a, b, accg[j]);
        }
      }
      __syncthreads();
#pragma unroll
      for (int j = 0; j < 4; ++j) {
        int nt = wave * 4 + j;
        int col = nt * 16 + (lane & 15), r0 = (lane >> 4) * 8;
#pragma unroll
        for (int i = 0; i < 8; ++i) {
          int r = r0 + i;
          float zv = (float)sZ[r * 512 + col];
          sZ[r * 512 + col] = (_Float16)(zv * sigm(accg[j][i]));
        }
      }
    }
    __syncthreads();

    // -- phase 7: LSTM gates g = z*Wih^T + h*Whh^T  (+bias in phase 8)
    {
      v8f accl[4];
#pragma unroll
      for (int j = 0; j < 4; ++j) {
        int nt = wave * 4 + j;
        accl[j] = (v8f){};
#pragma unroll
        for (int kc = 0; kc < 512; kc += 32) {
          v16h a = ld_tile(sZ + kc, 512, lane);
          v16h b = ld_tile(Wih + (size_t)nt * 16 * 512 + kc, 512, lane);
          accl[j] = WMMA_F16(a, b, accl[j]);
        }
      }
      __syncthreads();
      for (int i = tid; i < 2048; i += 256) sZ[i] = (_Float16)sh[i]; // h f16
      __syncthreads();
#pragma unroll
      for (int j = 0; j < 4; ++j) {
        int nt = wave * 4 + j;
#pragma unroll
        for (int kc = 0; kc < 128; kc += 32) {
          v16h a = ld_tile(sZ + kc, 128, lane);
          v16h b = ld_tile(Whh + (size_t)nt * 16 * 128 + kc, 128, lane);
          accl[j] = WMMA_F16(a, b, accl[j]);
        }
        st_tile_f32(sScr + nt * 16, 512, lane, accl[j]);
      }
    }
    __syncthreads();

    // -- phase 8: LSTM cell elementwise (i,f,g,o)
    for (int e = tid; e < 2048; e += 256) {
      int b = e >> 7, c = e & 127;
      const float* g = sScr + b * 512;
      float i_ = sigm(g[c] + bias[c]);
      float f_ = sigm(g[c + 128] + bias[c + 128]);
      float gg = tanhf(g[c + 256] + bias[c + 256]);
      float o_ = sigm(g[c + 384] + bias[c + 384]);
      float c2 = f_ * sc_[e] + i_ * gg;
      float h2 = o_ * tanhf(c2);
      sc_[e] = c2;
      sh[e] = h2;
      size_t o = (size_t)tt * 4096 + (size_t)b * 256 + dir * 128 + c;
      Hout[o] = h2;
      Hout16[o] = (_Float16)h2;
    }
    __syncthreads();
  }
}

// ---------------------------------------------------------------------------
// Answer pointer network: 2 sequential steps, tiny — pure fp32 VALU, 1 block.
// att = Hs@Wal^T precomputed with the WMMA GEMM. LDS: 64 KB.
// ---------------------------------------------------------------------------
__global__ __launch_bounds__(256) void k_pointer(
    const float* __restrict__ Hq,   // [48,16,256]
    const float* __restrict__ Hs,   // [256,16,256]
    const float* __restrict__ AttP, // [4096,128]
    const float* __restrict__ Waa, const float* __restrict__ w_beta,
    const float* __restrict__ Wah, const float* __restrict__ bah,
    const float* __restrict__ Wac, const float* __restrict__ bac,
    const float* __restrict__ ptWih, const float* __restrict__ ptWhh,
    const float* __restrict__ ptb,
    float* __restrict__ out) {       // [2,256,16]
  __shared__ float sh[2048], sc_[2048], sw[4096], R[8192];
  int tid = threadIdx.x;

  for (int e = tid; e < 4096; e += 256) { // qp = mean_q Hq
    int b = e >> 8, d = e & 255;
    float s = 0.0f;
    for (int q = 0; q < 48; ++q) s += Hq[((size_t)q * 16 + b) * 256 + d];
    sw[e] = s * (1.0f / 48.0f);
  }
  __syncthreads();
  for (int e = tid; e < 2048; e += 256) { // h,c init from pooled question
    int b = e >> 7, c = e & 127;
    float a = bah[c], cc = bac[c];
    for (int d = 0; d < 256; ++d) {
      float q = sw[b * 256 + d];
      a  += q * Wah[(size_t)c * 256 + d];
      cc += q * Wac[(size_t)c * 256 + d];
    }
    sh[e] = a; sc_[e] = cc;
  }
  __syncthreads();

  for (int step = 0; step < 2; ++step) {
    for (int e = tid; e < 2048; e += 256) { // proj = h@Waa^T
      int b = e >> 7, c = e & 127;
      float s = 0.0f;
      for (int d = 0; d < 128; ++d) s += sh[b * 128 + d] * Waa[(size_t)c * 128 + d];
      R[e] = s;
    }
    __syncthreads();
    for (int e = tid; e < 4096; e += 256) { // scores
      int b = e & 15;
      const float* am = AttP + (size_t)e * 128;
      const float* pr = R + b * 128;
      float s = 0.0f;
      for (int c = 0; c < 128; ++c) s += w_beta[c] * tanhf(am[c] + pr[c]);
      R[2048 + e] = s;
    }
    __syncthreads();
    if (tid < 16) { // softmax over passage
      int b = tid;
      float m = -1e30f;
      for (int p = 0; p < 256; ++p) m = fmaxf(m, R[2048 + p * 16 + b]);
      float s = 0.0f;
      for (int p = 0; p < 256; ++p) {
        float ex = expf(R[2048 + p * 16 + b] - m);
        R[2048 + p * 16 + b] = ex; s += ex;
      }
      float inv = 1.0f / s;
      for (int p = 0; p < 256; ++p) R[2048 + p * 16 + b] *= inv;
    }
    __syncthreads();
    for (int e = tid; e < 4096; e += 256) out[(size_t)step * 4096 + e] = R[2048 + e];
    for (int e = tid; e < 4096; e += 256) { // wv = beta-weighted Hs
      int b = e >> 8, d = e & 255;
      float s = 0.0f;
      for (int p = 0; p < 256; ++p)
        s += R[2048 + p * 16 + b] * Hs[((size_t)p * 16 + b) * 256 + d];
      sw[e] = s;
    }
    __syncthreads();
    for (int e = tid; e < 8192; e += 256) { // LSTM gates
      int b = e >> 9, c = e & 511;
      float s = ptb[c];
      for (int d = 0; d < 256; ++d) s += sw[b * 256 + d] * ptWih[(size_t)c * 256 + d];
      for (int d = 0; d < 128; ++d) s += sh[b * 128 + d] * ptWhh[(size_t)c * 128 + d];
      R[e] = s;
    }
    __syncthreads();
    for (int e = tid; e < 2048; e += 256) { // cell update
      int b = e >> 7, c = e & 127;
      const float* g = R + b * 512;
      float i_ = sigm(g[c]), f_ = sigm(g[c + 128]);
      float gg = tanhf(g[c + 256]), o_ = sigm(g[c + 384]);
      float c2 = f_ * sc_[e] + i_ * gg;
      sh[e] = o_ * tanhf(c2);
      sc_[e] = c2;
    }
    __syncthreads();
  }
}

// ---------------------------------------------------------------------------
// Host orchestration.
// ---------------------------------------------------------------------------
extern "C" void kernel_launch(void* const* d_in, const int* in_sizes, int n_in,
                              void* d_out, int out_size, void* d_ws, size_t ws_size,
                              hipStream_t stream) {
  (void)in_sizes; (void)n_in; (void)out_size; (void)ws_size;
  const float* p_inp  = (const float*)d_in[0];
  const float* q_inp  = (const float*)d_in[1];
  const float* g0_Wih = (const float*)d_in[2];
  const float* g0_Whh = (const float*)d_in[3];
  const float* g0_bih = (const float*)d_in[4];
  const float* g0_bhh = (const float*)d_in[5];
  const float* g1_Wih = (const float*)d_in[6];
  const float* g1_Whh = (const float*)d_in[7];
  const float* g1_bih = (const float*)d_in[8];
  const float* g1_bhh = (const float*)d_in[9];
  const float* g2_Wih = (const float*)d_in[10];
  const float* g2_Whh = (const float*)d_in[11];
  const float* g2_bih = (const float*)d_in[12];
  const float* g2_bhh = (const float*)d_in[13];
  const float* Wq     = (const float*)d_in[14];
  const float* Wp     = (const float*)d_in[15];
  const float* Wh     = (const float*)d_in[16];
  const float* w_alpha= (const float*)d_in[17];
  const float* Wg_m   = (const float*)d_in[18];
  const float* m_Wih  = (const float*)d_in[19];
  const float* m_Whh  = (const float*)d_in[20];
  const float* m_b    = (const float*)d_in[21];
  const float* Wsp    = (const float*)d_in[22];
  const float* Wsh    = (const float*)d_in[23];
  const float* w_gamma= (const float*)d_in[24];
  const float* Wg_s   = (const float*)d_in[25];
  const float* s_Wih  = (const float*)d_in[26];
  const float* s_Whh  = (const float*)d_in[27];
  const float* s_b    = (const float*)d_in[28];
  const float* Wal    = (const float*)d_in[29];
  const float* Waa    = (const float*)d_in[30];
  const float* w_beta = (const float*)d_in[31];
  const float* Wah    = (const float*)d_in[32];
  const float* bah    = (const float*)d_in[33];
  const float* Wac    = (const float*)d_in[34];
  const float* bac    = (const float*)d_in[35];
  const float* pt_Wih = (const float*)d_in[36];
  const float* pt_Whh = (const float*)d_in[37];
  const float* pt_b   = (const float*)d_in[38];

  char* base = (char*)d_ws;
  size_t off = 0;
  auto alloc = [&](size_t bytes) -> void* {
    void* p = (void*)(base + off);
    off += (bytes + 255) & ~(size_t)255;
    return p;
  };
  auto a16 = [&](size_t n) { return (_Float16*)alloc(2 * n); };
  auto af  = [&](size_t n) { return (float*)alloc(4 * n); };

  // f16 conversions (K padded to multiples of 32 where needed)
  _Float16* XP16  = a16((size_t)4096 * 576);
  _Float16* XQ16  = a16((size_t)768 * 576);
  _Float16* W0IH  = a16((size_t)384 * 576);
  _Float16* WHH0  = a16((size_t)384 * 128);
  _Float16* WIH1  = a16((size_t)384 * 128);
  _Float16* WHH1  = a16((size_t)384 * 128);
  _Float16* WIH2  = a16((size_t)384 * 128);
  _Float16* WHH2  = a16((size_t)384 * 128);
  _Float16* WQ16  = a16((size_t)128 * 256);
  _Float16* WP16  = a16((size_t)128 * 256);
  _Float16* WH16  = a16((size_t)128 * 128);
  _Float16* WGM16 = a16((size_t)512 * 512);
  _Float16* MWIH16= a16((size_t)512 * 512);
  _Float16* MWHH16= a16((size_t)512 * 128);
  _Float16* WSP16 = a16((size_t)128 * 256);
  _Float16* WSH16 = a16((size_t)128 * 128);
  _Float16* WGS16 = a16((size_t)512 * 512);
  _Float16* SWIH16= a16((size_t)512 * 512);
  _Float16* SWHH16= a16((size_t)512 * 128);
  _Float16* WAL16 = a16((size_t)128 * 256);
  _Float16* HP16  = a16((size_t)4096 * 256);
  _Float16* HQ16  = a16((size_t)768 * 256);
  _Float16* HR16  = a16((size_t)4096 * 256);
  _Float16* HS16  = a16((size_t)4096 * 256);

  // f32 intermediates
  float* GP0  = af((size_t)4096 * 384);
  float* GQ0  = af((size_t)768 * 384);
  float* HP   = af((size_t)4096 * 256);
  float* HQ   = af((size_t)768 * 256);
  float* HR   = af((size_t)4096 * 256);
  float* HS   = af((size_t)4096 * 256);
  float* ATTQ = af((size_t)768 * 128);
  float* ATTR = af((size_t)4096 * 128);
  float* ATTP = af((size_t)4096 * 128);

  auto cast = [&](const float* src, _Float16* dst, int rows, int K, int Kpad) {
    size_t n = (size_t)rows * Kpad;
    int blocks = (int)((n + 255) / 256);
    if (blocks > 4096) blocks = 4096;
    k_cast_pad<<<blocks, 256, 0, stream>>>(src, dst, rows, K, Kpad);
  };
  cast(p_inp, XP16, 4096, 556, 576);
  cast(q_inp, XQ16, 768, 556, 576);
  cast(g0_Wih, W0IH, 384, 556, 576);
  cast(g0_Whh, WHH0, 384, 128, 128);
  cast(g1_Wih, WIH1, 384, 128, 128);
  cast(g1_Whh, WHH1, 384, 128, 128);
  cast(g2_Wih, WIH2, 384, 128, 128);
  cast(g2_Whh, WHH2, 384, 128, 128);
  cast(Wq, WQ16, 128, 256, 256);
  cast(Wp, WP16, 128, 256, 256);
  cast(Wh, WH16, 128, 128, 128);
  cast(Wg_m, WGM16, 512, 512, 512);
  cast(m_Wih, MWIH16, 512, 512, 512);
  cast(m_Whh, MWHH16, 512, 128, 128);
  cast(Wsp, WSP16, 128, 256, 256);
  cast(Wsh, WSH16, 128, 128, 128);
  cast(Wg_s, WGS16, 512, 512, 512);
  cast(s_Wih, SWIH16, 512, 512, 512);
  cast(s_Whh, SWHH16, 512, 128, 128);
  cast(Wal, WAL16, 128, 256, 256);

  // Time-batched layer-0 GRU input projections (WMMA GEMM)
  k_gemm_wmma<<<dim3(3, 256), 256, 0, stream>>>(XP16, 576, W0IH, 576, g0_bih,
                                                GP0, nullptr, 384, 576);
  k_gemm_wmma<<<dim3(3, 48), 256, 0, stream>>>(XQ16, 576, W0IH, 576, g0_bih,
                                               GQ0, nullptr, 384, 576);

  // Recurrent 3-layer GRU, both directions concurrently (blockIdx.x = dir)
  k_gru_dir<<<2, 256, 0, stream>>>(GP0, g0_bhh, WHH0, WIH1, WHH1, g1_bih, g1_bhh,
                                   WIH2, WHH2, g2_bih, g2_bhh, HP, HP16, 256);
  k_gru_dir<<<2, 256, 0, stream>>>(GQ0, g0_bhh, WHH0, WIH1, WHH1, g1_bih, g1_bhh,
                                   WIH2, WHH2, g2_bih, g2_bhh, HQ, HQ16, 48);

  // Match-LSTM: att_mem = Hq @ Wq^T, then gated attention LSTM over Hp
  k_gemm_wmma<<<dim3(1, 48), 256, 0, stream>>>(HQ16, 256, WQ16, 256, nullptr,
                                               ATTQ, nullptr, 128, 256);
  k_attn_lstm<<<2, 256, 0, stream>>>(HP16, HQ, ATTQ, WP16, WH16, w_alpha,
                                     WGM16, MWIH16, MWHH16, m_b, HR, HR16, 256, 48);

  // Self-matching: att_mem = Hr @ Wsp^T, gated attention LSTM over Hr
  k_gemm_wmma<<<dim3(1, 256), 256, 0, stream>>>(HR16, 256, WSP16, 256, nullptr,
                                                ATTR, nullptr, 128, 256);
  k_attn_lstm<<<2, 256, 0, stream>>>(HR16, HR, ATTR, WSP16, WSH16, w_gamma,
                                     WGS16, SWIH16, SWHH16, s_b, HS, HS16, 256, 256);

  // Pointer network: att = Hs @ Wal^T then 2 tiny sequential steps
  k_gemm_wmma<<<dim3(1, 256), 256, 0, stream>>>(HS16, 256, WAL16, 256, nullptr,
                                                ATTP, nullptr, 128, 256);
  k_pointer<<<1, 256, 0, stream>>>(HQ, HS, ATTP, Waa, w_beta, Wah, bah, Wac, bac,
                                   pt_Wih, pt_Whh, pt_b, (float*)d_out);
  (void)HP;
}
